// HashSideOut_54357106098900
// MI455X (gfx1250) — matrix operand order repaired and checked by reading, not compiled, not run on hardware
//
#include <hip/hip_runtime.h>
#include <hip/hip_bf16.h>
#include <math.h>

typedef __attribute__((ext_vector_type(16))) _Float16 v16h;
typedef __attribute__((ext_vector_type(8)))  _Float16 v8h;
typedef __attribute__((ext_vector_type(2)))  _Float16 v2h;
typedef __attribute__((ext_vector_type(8)))  float    v8f;

#define BATCH  8
#define LVLS   16
#define TBL    65536
#define NPIX   65536        // 256*256
#define SDIM   512
#define HPRIME 2654435761u
#define FSTR   40           // LDS feature row stride in halves (bank spread)

// Instant-NGP resolution progression: round(16 * (256/16)^(i/15))
__constant__ int kRes[LVLS] = {16,19,23,28,34,40,49,58,70,84,102,122,147,177,213,256};

__device__ __forceinline__ v8f wmma16x16x32(v16h a, v16h b, v8f c) {
    // D = A(16x32 f16) x B(32x16 f16) + C(16x16 f32)
    return __builtin_amdgcn_wmma_f32_16x16x32_f16(
        /*neg_a=*/false, a, /*neg_b=*/false, b,
        /*c_mod=*/(short)0, c, /*reuse_a=*/false, /*reuse_b=*/false);
}

// ---------------------------------------------------------------------------
// Kernel 1: per (batch, layer) effective modulated weights (f16, demod folded)
//   style[i] = s[b] . aw[i] + ab[i]          (cin == 32 for all 3 layers)
//   Weff[o][i] = w[o][i]*style[i]*rsqrt(sum_i (w[o][i]*style[i])^2 + 1e-8)
// Layer2 (cout=3) padded to 16 rows of zeros for the WMMA B operand.
// ---------------------------------------------------------------------------
__global__ void HashSideOut_modweights(
    const float* __restrict__ s,
    const float* __restrict__ w0, const float* __restrict__ aw0, const float* __restrict__ ab0,
    const float* __restrict__ w1, const float* __restrict__ aw1, const float* __restrict__ ab1,
    const float* __restrict__ w2, const float* __restrict__ aw2, const float* __restrict__ ab2,
    _Float16* __restrict__ weff0, _Float16* __restrict__ weff1, _Float16* __restrict__ weff2)
{
    const int b = blockIdx.x, layer = blockIdx.y, tid = threadIdx.x; // 32 threads
    const float *w, *aw, *ab; _Float16* dst; int cout, rows;
    if (layer == 0)      { w = w0; aw = aw0; ab = ab0; dst = weff0; cout = 32; rows = 32; }
    else if (layer == 1) { w = w1; aw = aw1; ab = ab1; dst = weff1; cout = 32; rows = 32; }
    else                 { w = w2; aw = aw2; ab = ab2; dst = weff2; cout = 3;  rows = 16; }

    __shared__ float style[32];
    {
        const float* sv = s + (size_t)b * SDIM;
        const float* av = aw + (size_t)tid * SDIM;
        float acc = 0.f;
        for (int d = 0; d < SDIM; ++d) acc += sv[d] * av[d];
        style[tid] = acc + ab[tid];
    }
    __syncthreads();

    if (tid < rows) {
        _Float16* drow = dst + ((size_t)b * rows + tid) * 32;
        if (tid < cout) {
            const float* wrow = w + (size_t)tid * 32;
            float wm[32]; float ss = 0.f;
            #pragma unroll
            for (int i = 0; i < 32; ++i) { wm[i] = wrow[i] * style[i]; ss += wm[i] * wm[i]; }
            const float demod = rsqrtf(ss + 1e-8f);
            #pragma unroll
            for (int i = 0; i < 32; ++i) drow[i] = (_Float16)(wm[i] * demod);
        } else {
            #pragma unroll
            for (int i = 0; i < 32; ++i) drow[i] = (_Float16)0.f;
        }
    }
}

// ---------------------------------------------------------------------------
// Kernel 2: fused hash-grid gather + 3-layer modulated MLP on WMMA.
// Block = 256 threads = 8 waves; each wave owns a 16-pixel tile of one batch.
// ---------------------------------------------------------------------------
__global__ void __launch_bounds__(256) HashSideOut_hashmlp(
    const float* __restrict__ x, const float* __restrict__ coords,
    const _Float16* __restrict__ weff0, const _Float16* __restrict__ weff1,
    const _Float16* __restrict__ weff2,
    const float* __restrict__ b0, const float* __restrict__ b1, const float* __restrict__ b2,
    float* __restrict__ out)
{
    __shared__ _Float16 feat[8][16 * FSTR];   // per-wave 16x32 f16 tile (stride-40 rows)

    const int b    = blockIdx.y;
    const int wave = threadIdx.x >> 5;
    const int lane = threadIdx.x & 31;
    const int m    = lane & 15;               // pixel-in-tile / WMMA column id
    const int grp  = lane >> 4;               // lane half (0/1)
    const int rowBase = (blockIdx.x * 8 + wave) * 16;  // first pixel of this tile
    _Float16* fw = feat[wave];

    // ---- Phase 1: hash-grid gather. Lane handles pixel (rowBase+m), levels grp*8..grp*8+7
    {
        const int n = rowBase + m;
        const float cx = coords[2 * n + 0];
        const float cy = coords[2 * n + 1];
        const float* xb = x + (size_t)b * ((size_t)LVLS * TBL * 2);
        #pragma unroll
        for (int j = 0; j < 8; ++j) {
            const int lev = grp * 8 + j;
            const int R   = kRes[lev];
            const float rx = (float)R - 1.0f;
            const float px = cx * rx, py = cy * rx;
            const float fx = floorf(px), fy = floorf(py);
            const float wx = px - fx,  wy = py - fy;
            const unsigned ix0 = (unsigned)fx, iy0 = (unsigned)fy;
            const unsigned mx  = (unsigned)(R - 1);
            const unsigned ix1 = (ix0 + 1u < mx) ? ix0 + 1u : mx;
            const unsigned iy1 = (iy0 + 1u < mx) ? iy0 + 1u : mx;
            const float2* tb = (const float2*)(xb + (size_t)lev * (TBL * 2));
            const unsigned h00 = (ix0 ^ (iy0 * HPRIME)) & 0xFFFFu;
            const unsigned h10 = (ix1 ^ (iy0 * HPRIME)) & 0xFFFFu;
            const unsigned h01 = (ix0 ^ (iy1 * HPRIME)) & 0xFFFFu;
            const unsigned h11 = (ix1 ^ (iy1 * HPRIME)) & 0xFFFFu;
            const float2 g00 = tb[h00], g10 = tb[h10], g01 = tb[h01], g11 = tb[h11];
            const float w00 = (1.f - wx) * (1.f - wy), w10 = wx * (1.f - wy);
            const float w01 = (1.f - wx) * wy,         w11 = wx * wy;
            const float f0 = g00.x * w00 + g10.x * w10 + g01.x * w01 + g11.x * w11;
            const float f1 = g00.y * w00 + g10.y * w10 + g01.y * w01 + g11.y * w11;
            v2h p; p.x = (_Float16)f0; p.y = (_Float16)f1;
            *(v2h*)&fw[m * FSTR + 2 * lev] = p;
        }
    }
    __syncthreads();

    union U16 { v16h v; v8h h[2]; };

    // A operand (16x32 f16): lane<16 holds K 0..7 & 16..23; lane>=16 holds K 8..15 & 24..31
    auto loadA = [&]() -> v16h {
        U16 a;
        const v8h* rp = (const v8h*)&fw[m * FSTR];
        a.h[0] = rp[grp];          // K  grp*8 .. grp*8+7
        a.h[1] = rp[2 + grp];      // K 16+grp*8 .. 16+grp*8+7
        return a.v;
    };
    // B operand (32x16 f16): lane holds column (colOff+m), K rows grp*8..grp*8+7 & 16+grp*8..16+grp*8+7
    auto loadB = [&](const _Float16* wt, int colOff) -> v16h {
        U16 bm;
        const v8h* rp = (const v8h*)(wt + (size_t)(colOff + m) * 32);
        bm.h[0] = rp[grp];
        bm.h[1] = rp[2 + grp];
        return bm.v;
    };

    const _Float16* W0 = weff0 + (size_t)b * 32 * 32;
    const _Float16* W1 = weff1 + (size_t)b * 32 * 32;
    const _Float16* W2 = weff2 + (size_t)b * 16 * 32;
    const int rb = grp * 8;  // D row base for this lane half

    // ---- Layer 0: 32 -> 32, relu (bias pre-loaded into C)
    {
        v16h a = loadA();
        v8f c0, c1;
        const float bv0 = b0[m], bv1 = b0[16 + m];
        #pragma unroll
        for (int r = 0; r < 8; ++r) { c0[r] = bv0; c1[r] = bv1; }
        c0 = wmma16x16x32(a, loadB(W0, 0),  c0);
        c1 = wmma16x16x32(a, loadB(W0, 16), c1);
        #pragma unroll
        for (int r = 0; r < 8; ++r) {
            fw[(rb + r) * FSTR + m]      = (_Float16)fmaxf(c0[r], 0.f);
            fw[(rb + r) * FSTR + 16 + m] = (_Float16)fmaxf(c1[r], 0.f);
        }
    }
    __syncthreads();

    // ---- Layer 1: 32 -> 32, relu
    {
        v16h a = loadA();
        v8f c0, c1;
        const float bv0 = b1[m], bv1 = b1[16 + m];
        #pragma unroll
        for (int r = 0; r < 8; ++r) { c0[r] = bv0; c1[r] = bv1; }
        c0 = wmma16x16x32(a, loadB(W1, 0),  c0);
        c1 = wmma16x16x32(a, loadB(W1, 16), c1);
        #pragma unroll
        for (int r = 0; r < 8; ++r) {
            fw[(rb + r) * FSTR + m]      = (_Float16)fmaxf(c0[r], 0.f);
            fw[(rb + r) * FSTR + 16 + m] = (_Float16)fmaxf(c1[r], 0.f);
        }
    }
    __syncthreads();

    // ---- Layer 2: 32 -> 3 (padded to 16), tanh, NCHW store
    {
        v16h a = loadA();
        v8f c2;
        const float bv = (m < 3) ? b2[m] : 0.f;
        #pragma unroll
        for (int r = 0; r < 8; ++r) c2[r] = bv;
        c2 = wmma16x16x32(a, loadB(W2, 0), c2);
        if (m < 3) {
            float* op = out + ((size_t)(b * 3 + m)) * NPIX + rowBase + rb;
            #pragma unroll
            for (int r = 0; r < 8; ++r) op[r] = tanhf(c2[r]);
        }
    }
}

// ---------------------------------------------------------------------------
extern "C" void kernel_launch(void* const* d_in, const int* in_sizes, int n_in,
                              void* d_out, int out_size, void* d_ws, size_t ws_size,
                              hipStream_t stream)
{
    (void)in_sizes; (void)n_in; (void)out_size; (void)ws_size;
    const float* x      = (const float*)d_in[0];
    const float* s      = (const float*)d_in[1];
    const float* coords = (const float*)d_in[2];
    const float* w0  = (const float*)d_in[3];
    const float* aw0 = (const float*)d_in[4];
    const float* ab0 = (const float*)d_in[5];
    const float* b0  = (const float*)d_in[6];
    const float* w1  = (const float*)d_in[7];
    const float* aw1 = (const float*)d_in[8];
    const float* ab1 = (const float*)d_in[9];
    const float* b1  = (const float*)d_in[10];
    const float* w2  = (const float*)d_in[11];
    const float* aw2 = (const float*)d_in[12];
    const float* ab2 = (const float*)d_in[13];
    const float* b2  = (const float*)d_in[14];
    float* out = (float*)d_out;

    _Float16* weff0 = (_Float16*)d_ws;                 // [B][32][32] f16
    _Float16* weff1 = weff0 + (size_t)BATCH * 32 * 32; // [B][32][32] f16
    _Float16* weff2 = weff1 + (size_t)BATCH * 32 * 32; // [B][16][32] f16 (rows 3..15 zero)

    HashSideOut_modweights<<<dim3(BATCH, 3), 32, 0, stream>>>(
        s, w0, aw0, ab0, w1, aw1, ab1, w2, aw2, ab2, weff0, weff1, weff2);

    HashSideOut_hashmlp<<<dim3(NPIX / 128, BATCH), 256, 0, stream>>>(
        x, coords, weff0, weff1, weff2, b0, b1, b2, out);
}
